// RNN_echostate_37417755083264
// MI455X (gfx1250) — matrix operand order repaired and checked by reading, not compiled, not run on hardware
//
#include <hip/hip_runtime.h>
#include <cmath>

typedef unsigned short u16;
typedef __bf16 bf16_t;
typedef __attribute__((ext_vector_type(16))) bf16_t v16bf;
typedef __attribute__((ext_vector_type(16))) u16    v16u;
typedef __attribute__((ext_vector_type(8)))  u16    v8u;
typedef __attribute__((ext_vector_type(8)))  float  v8f;

#define B_    64
#define T_    2048
#define NIN   64
#define NREC  1024
#define KTOT  1088            // NIN + NREC (fused GEMM K)
#define KCH   (KTOT / 32)     // 34 bf16 WMMA K-chunks
#define NOISE_STD 0.001f
#define NWG   32
#define WAVES 8               // waves per WG (256 threads, wave32)

// workspace layout (bytes)
#define WS_WCAT   0u
#define WS_ABUF0  2228224u               // 1024*1088*2
#define WS_ABUF1  (WS_ABUF0 + 131072u)   // 64*1024*2
#define WS_XBF    (WS_ABUF1 + 131072u)
#define WS_CTR    (WS_XBF + 16777216u)   // 64*2048*64*2

// f32 -> bf16 bits, round-to-nearest-even
__device__ __forceinline__ u16 f2bf(float f) {
  unsigned u = __builtin_bit_cast(unsigned, f);
  unsigned r = u + 0x7fffu + ((u >> 16) & 1u);
  return (u16)(r >> 16);
}

// fast tanh: 1 - 2/(1 + 2^(x*2/ln2)); saturates correctly to +-1 via inf/0
__device__ __forceinline__ float fast_tanh(float x) {
  float e2 = __builtin_amdgcn_exp2f(x * 2.885390081777927f);
  return 1.0f - 2.0f * __builtin_amdgcn_rcpf(e2 + 1.0f);
}

// Prep (re-run every call -> deterministic): pack [W_ih|W_hh] to bf16,
// pre-convert x to bf16, zero t=0 hidden staging, reset barrier counter.
__global__ void rnn_prep(const float* __restrict__ x,
                         const float* __restrict__ Wih,
                         const float* __restrict__ Whh,
                         u16* __restrict__ Wcat,
                         u16* __restrict__ Abuf0,
                         u16* __restrict__ xbf,
                         unsigned* __restrict__ counter)
{
  int idx = blockIdx.x * blockDim.x + threadIdx.x;
  if (idx < B_ * T_ * NIN) xbf[idx] = f2bf(x[idx]);
  if (idx < NREC * KTOT) {
    int r = idx / KTOT;
    int k = idx - r * KTOT;
    float v = (k < NIN) ? Wih[r * NIN + k] : Whh[r * NREC + (k - NIN)];
    Wcat[idx] = f2bf(v);
  }
  if (idx < B_ * NREC) Abuf0[idx] = 0;   // bf16 zero (tanh(h0)=0)
  if (idx == 0) *counter = 0u;
}

// Persistent cooperative scan: 32 WGs x 8 waves = 256 waves, each wave owns
// one 16x16 output tile for all 2048 steps. Its full B slice (34 fragments,
// 272 VGPRs) is held in registers for the entire kernel.
__global__ __launch_bounds__(256, 1) void rnn_scan_persistent(
    const u16*   __restrict__ xbf,    // [B, T, NIN] bf16
    const float* __restrict__ noise,  // [T, NREC]
    const u16*   __restrict__ Wcat,   // [NREC, KTOT] bf16
    u16* __restrict__ Abuf0,          // [B, NREC] bf16 (double buffer)
    u16* __restrict__ Abuf1,
    unsigned* __restrict__ counter,
    float* __restrict__ out,          // [NREC, T]
    float* __restrict__ hlast)        // [B, NREC]
{
  const int tid  = threadIdx.x;
  const int lane = tid & 31;
  const int wv   = tid >> 5;
  const int tile = blockIdx.x * WAVES + wv;
  const int nt   = tile & 63;   // N tile (output units)
  const int mt   = tile >> 6;   // M tile (batch)
  const int nloc = lane & 15;
  const int hi   = lane >> 4;

  const int mrow = mt * 16 + nloc;               // A-fragment row (batch)
  const int rrow = nt * 16 + nloc;               // B-fragment row (unit r)

  // ---- hoist this wave's entire B slice into registers (34 x v16u)
  const u16* wrow = Wcat + (size_t)rrow * KTOT;
  v16u breg[KCH];
  #pragma unroll
  for (int kc = 0; kc < KCH; ++kc)
  {
    v8u blo = *(const v8u*)(wrow + kc * 32 + hi * 8);
    v8u bhi = *(const v8u*)(wrow + kc * 32 + hi * 8 + 16);
    #pragma unroll
    for (int e = 0; e < 8; ++e) { breg[kc][e] = blo[e]; breg[kc][e + 8] = bhi[e]; }
  }

  const u16* xrow = xbf + (size_t)mrow * T_ * NIN;

  for (int t = 0; t < T_; ++t) {
    const u16* Ac = (t & 1) ? Abuf1 : Abuf0;
    u16*       An = (t & 1) ? Abuf0 : Abuf1;

    v8f acc = {};

    // ---- input projection: K = 0..63 from pre-converted bf16 x_t
    const u16* xt = xrow + t * NIN;
    #pragma unroll
    for (int kc = 0; kc < 2; ++kc) {
      const int kb = kc * 32 + hi * 8;
      v8u alo = *(const v8u*)(xt + kb);
      v8u ahi = *(const v8u*)(xt + kb + 16);
      v16u af;
      #pragma unroll
      for (int e = 0; e < 8; ++e) { af[e] = alo[e]; af[e + 8] = ahi[e]; }
      acc = __builtin_amdgcn_wmma_f32_16x16x32_bf16(
          false, __builtin_bit_cast(v16bf, af),
          false, __builtin_bit_cast(v16bf, breg[kc]),
          (short)0, acc, false, false);
    }

    // ---- recurrent: K = 0..1023 over staged tanh(h_prev), B from registers
    const u16* arow = Ac + (size_t)mrow * NREC;
    #pragma unroll
    for (int kc = 0; kc < 32; ++kc) {
      const int kb = kc * 32 + hi * 8;
      v8u alo = *(const v8u*)(arow + kb);        // K = kb .. kb+7
      v8u ahi = *(const v8u*)(arow + kb + 16);   // K = kb+16 .. kb+23
      v16u af;
      #pragma unroll
      for (int e = 0; e < 8; ++e) { af[e] = alo[e]; af[e + 8] = ahi[e]; }
      acc = __builtin_amdgcn_wmma_f32_16x16x32_bf16(
          false, __builtin_bit_cast(v16bf, af),
          false, __builtin_bit_cast(v16bf, breg[kc + 2]),
          (short)0, acc, false, false);
    }

    // ---- epilogue: noise, outputs, stage tanh(h) as next-step A (bf16)
    const float nz = noise[(size_t)t * NREC + rrow] * NOISE_STD;
    #pragma unroll
    for (int j = 0; j < 8; ++j) {
      const int mg = mt * 16 + j + 8 * hi;       // C/D layout: M = j + 8*hi
      float h = acc[j] + nz;
      An[(size_t)mg * NREC + rrow] = f2bf(fast_tanh(h));
      if (mg == B_ - 1) out[(size_t)rrow * T_ + t] = h;   // h_t[-1]
      if (t == T_ - 1)  hlast[(size_t)mg * NREC + rrow] = h;
    }

    // ---- grid barrier: agent fence by all, then 1 signal + poll per WG
    __threadfence();                     // agent-scope release of our stores
    __syncthreads();
    if (tid == 0) {
      __hip_atomic_fetch_add(counter, 1u, __ATOMIC_RELAXED,
                             __HIP_MEMORY_SCOPE_AGENT);
      const unsigned target = (unsigned)NWG * (unsigned)(t + 1);
      while (__hip_atomic_load(counter, __ATOMIC_ACQUIRE,
                               __HIP_MEMORY_SCOPE_AGENT) < target)
        __builtin_amdgcn_s_sleep(1);     // acquire inv covers this WGP's L0
    }
    __syncthreads();
  }
}

extern "C" void kernel_launch(void* const* d_in, const int* in_sizes, int n_in,
                              void* d_out, int out_size, void* d_ws, size_t ws_size,
                              hipStream_t stream)
{
  (void)in_sizes; (void)n_in; (void)out_size; (void)ws_size;
  const float* x     = (const float*)d_in[0];
  const float* Wih   = (const float*)d_in[1];
  const float* Whh   = (const float*)d_in[2];
  const float* noise = (const float*)d_in[3];

  float* out   = (float*)d_out;
  float* hlast = out + (size_t)NREC * T_;

  char* ws = (char*)d_ws;
  u16* Wcat  = (u16*)(ws + WS_WCAT);
  u16* Abuf0 = (u16*)(ws + WS_ABUF0);
  u16* Abuf1 = (u16*)(ws + WS_ABUF1);
  u16* xbf   = (u16*)(ws + WS_XBF);
  unsigned* counter = (unsigned*)(ws + WS_CTR);

  const int prepN = B_ * T_ * NIN;                 // largest prep range
  hipLaunchKernelGGL(rnn_prep, dim3((prepN + 255) / 256), dim3(256), 0, stream,
                     x, Wih, Whh, Wcat, Abuf0, xbf, counter);
  hipLaunchKernelGGL(rnn_scan_persistent, dim3(NWG), dim3(256), 0, stream,
                     xbf, noise, Wcat, Abuf0, Abuf1, counter, out, hlast);
}
